// MultiHeadAttention_6236292514105
// MI455X (gfx1250) — matrix operand order repaired and checked by reading, not compiled, not run on hardware
//
#include <hip/hip_runtime.h>
#include <hip/hip_bf16.h>

#define B_    2
#define S_    2048
#define D_    2048
#define H_    32
#define HKV_  8
#define HD_   64
#define NREP_ (H_ / HKV_)
#define TOK_  (B_ * S_)
#define SCALE_ 0.125f  // HD^-0.5

typedef __attribute__((ext_vector_type(16))) __bf16 v16bf;
typedef __attribute__((ext_vector_type(8)))  float  v8f;

union Frag {
    v16bf v;
    uint4 q[2];
};

__device__ __forceinline__ v8f vzero8() {
    v8f z;
#pragma unroll
    for (int i = 0; i < 8; ++i) z[i] = 0.0f;
    return z;
}

__device__ __forceinline__ v8f wmma_bf16(Frag a, Frag b, v8f c) {
    // D = A(16x32 bf16) * B(32x16 bf16) + C(16x16 f32)
    return __builtin_amdgcn_wmma_f32_16x16x32_bf16(
        /*neg_a=*/false, a.v, /*neg_b=*/false, b.v,
        /*c_mod=*/(short)0, c, /*reuse_a=*/false, /*reuse_b=*/false);
}

// 16-byte async copy global -> LDS (CDNA5 GLOBAL_LOAD_ASYNC_TO_LDS_B128, ASYNCcnt)
__device__ __forceinline__ void async_copy_b128(const __bf16* gsrc, __bf16* ldst) {
    unsigned lds_off = (unsigned)(unsigned long long)ldst;  // generic addr low 32 bits = LDS offset
    unsigned long long gaddr = (unsigned long long)gsrc;
    asm volatile("global_load_async_to_lds_b128 %0, %1, off"
                 :: "v"(lds_off), "v"(gaddr) : "memory");
}

// ---------------------------------------------------------------------------
// Elementwise f32 -> bf16 cast
// ---------------------------------------------------------------------------
__global__ void cast_f32_bf16(const float* __restrict__ x, __bf16* __restrict__ y, long n) {
    long i = (long)blockIdx.x * blockDim.x + threadIdx.x;
    if (i < n) y[i] = (__bf16)x[i];
}

// ---------------------------------------------------------------------------
// W[K,N] f32 -> WT[N,K] bf16 (LDS-tiled transpose, both sides coalesced)
// ---------------------------------------------------------------------------
__global__ void transpose_cast(const float* __restrict__ W, __bf16* __restrict__ WT,
                               int K, int N) {
    __shared__ float tile[32][33];
    int n0 = blockIdx.x * 32;
    int k0 = blockIdx.y * 32;
    for (int i = threadIdx.y; i < 32; i += 8)
        tile[i][threadIdx.x] = W[(size_t)(k0 + i) * N + n0 + threadIdx.x];
    __syncthreads();
    for (int i = threadIdx.y; i < 32; i += 8)
        WT[(size_t)(n0 + i) * K + k0 + threadIdx.x] = (__bf16)tile[threadIdx.x][i];
}

// ---------------------------------------------------------------------------
// C[M,N] f32 = A[M,K] bf16 @ BT[N,K] bf16
// block: 256 threads = 8 waves; block tile 128(M) x 128(N); wave tile 32 x 64.
// B tile (128N x 32K = 8KB) staged in LDS via async-to-LDS, double buffered.
// ---------------------------------------------------------------------------
__global__ void __launch_bounds__(256)
gemm_bf16_f32(const __bf16* __restrict__ A, const __bf16* __restrict__ BT,
              float* __restrict__ C, int M, int N, int K) {
    __shared__ __align__(16) __bf16 bbuf[2][128 * 32];  // [buffer][n*32 + k], 16KB

    int tid  = threadIdx.x;
    int lane = tid & 31;
    int wave = tid >> 5;
    int col  = lane & 15;
    int half = lane >> 4;
    int mw   = wave & 3;   // M sub-block (32 rows each)
    int nw   = wave >> 2;  // N sub-block (64 cols each)

    int m0 = blockIdx.y * 128 + mw * 32;
    int n0 = blockIdx.x * 128;

    const __bf16* arow0 = A + (size_t)(m0 + col) * K;
    const __bf16* arow1 = A + (size_t)(m0 + 16 + col) * K;

    // async B-tile staging: each thread copies 2 x 16B; rows are BT rows (K-contig)
    int brow   = tid >> 1;            // 0..127
    int bchunk = (tid & 1) * 16;      // element offset within 32-K slab
    const __bf16* bsrc = BT + (size_t)(n0 + brow) * K + bchunk;
    __bf16* bdst0 = &bbuf[0][brow * 32 + bchunk];
    __bf16* bdst1 = &bbuf[1][brow * 32 + bchunk];

    v8f acc[2][4];
#pragma unroll
    for (int g = 0; g < 2; ++g)
#pragma unroll
        for (int f = 0; f < 4; ++f) acc[g][f] = vzero8();

    const int NT = K / 32;

    // prologue: stage tile 0
    async_copy_b128(bsrc + 0, bdst0);
    async_copy_b128(bsrc + 8, bdst0 + 8);  // second half of this thread's 32B? (row-major pair)

    for (int t = 0; t < NT; ++t) {
        int knext = (t + 1 < NT) ? (t + 1) * 32 : 0;  // clamped dummy issue on last iter
        __bf16* bdst = (((t + 1) & 1) ? bdst1 : bdst0);
        async_copy_b128(bsrc + knext, bdst);
        async_copy_b128(bsrc + knext + 8, bdst + 8);

        asm volatile("s_wait_asynccnt 0x2" ::: "memory");  // tile t landed (2 newest pending)
        __syncthreads();

        const __bf16* bt = bbuf[t & 1];
        int k0 = t * 32;

        Frag a0, a1;
        a0.q[0] = *(const uint4*)(arow0 + k0 + 8 * half);
        a0.q[1] = *(const uint4*)(arow0 + k0 + 16 + 8 * half);
        a1.q[0] = *(const uint4*)(arow1 + k0 + 8 * half);
        a1.q[1] = *(const uint4*)(arow1 + k0 + 16 + 8 * half);

#pragma unroll
        for (int f = 0; f < 4; ++f) {
            const __bf16* bp = bt + (size_t)(nw * 64 + 16 * f + col) * 32 + 8 * half;
            Frag b;
            b.q[0] = *(const uint4*)(bp);
            b.q[1] = *(const uint4*)(bp + 16);
            acc[0][f] = wmma_bf16(a0, b, acc[0][f]);
            acc[1][f] = wmma_bf16(a1, b, acc[1][f]);
        }
        __syncthreads();  // everyone done reading bbuf[t&1] before it is re-staged
    }

#pragma unroll
    for (int g = 0; g < 2; ++g)
#pragma unroll
        for (int f = 0; f < 4; ++f)
#pragma unroll
            for (int r = 0; r < 8; ++r)
                C[(size_t)(m0 + 16 * g + r + 8 * half) * N + n0 + nw * 64 + 16 * f + col] =
                    acc[g][f][r];
}

// ---------------------------------------------------------------------------
// RoPE + cast + layout change: X[B,S,nH*HD] f32 -> Out[B,nH,S,HD] bf16
// ---------------------------------------------------------------------------
__global__ void rope_cast(const float* __restrict__ X, __bf16* __restrict__ Out, int nH) {
    long idx   = (long)blockIdx.x * blockDim.x + threadIdx.x;
    long total = (long)B_ * S_ * nH * (HD_ / 2);
    if (idx >= total) return;
    int  i = idx % (HD_ / 2);
    long t = idx / (HD_ / 2);
    int  h = t % nH; t /= nH;
    int  s = t % S_;
    int  b = t / S_;

    float inv = __powf(10000.0f, -(2.0f * i) / (float)HD_);
    float ang = (float)s * inv;
    float sn, cs;
    __sincosf(ang, &sn, &cs);

    const float* xp = X + (((size_t)b * S_ + s) * nH + h) * HD_;
    float xe = xp[2 * i];
    float xo = xp[2 * i + 1];
    __bf16* op = Out + (((size_t)b * nH + h) * S_ + s) * HD_;
    op[2 * i]     = (__bf16)(xe * cs - xo * sn);
    op[2 * i + 1] = (__bf16)(xe * sn + xo * cs);
}

// ---------------------------------------------------------------------------
// V transpose + cast: X[B,S,HKV*HD] f32 -> VT[B,HKV,HD,S] bf16
// ---------------------------------------------------------------------------
__global__ void v_transpose_cast(const float* __restrict__ X, __bf16* __restrict__ VT) {
    __shared__ float tile[32][33];
    int bkv = blockIdx.z;
    int b   = bkv / HKV_;
    int kv  = bkv % HKV_;
    int s0  = blockIdx.x * 32;
    int d0  = blockIdx.y * 32;
    const float* xp = X + (size_t)b * S_ * (HKV_ * HD_) + (size_t)kv * HD_;
    for (int i = threadIdx.y; i < 32; i += 8)
        tile[i][threadIdx.x] = xp[(size_t)(s0 + i) * (HKV_ * HD_) + d0 + threadIdx.x];
    __syncthreads();
    __bf16* op = VT + ((size_t)b * HKV_ + kv) * (size_t)HD_ * S_;
    for (int i = threadIdx.y; i < 32; i += 8)
        op[(size_t)(d0 + i) * S_ + s0 + threadIdx.x] = (__bf16)tile[threadIdx.x][i];
}

// ---------------------------------------------------------------------------
// Flash attention (causal + attention_mask), GQA.
// Q[B,H,S,HD] bf16, K[B,HKV,S,HD] bf16, VT[B,HKV,HD,S] bf16 -> O[B,S,H,HD] bf16
// block = 128 threads = 4 waves; wave owns 16 query rows; key tiles of 32.
// ---------------------------------------------------------------------------
__global__ void __launch_bounds__(128)
flash_attn(const __bf16* __restrict__ Q, const __bf16* __restrict__ Kc,
           const __bf16* __restrict__ VT, const int* __restrict__ amask,
           __bf16* __restrict__ O) {
    __shared__ __align__(16) __bf16 pbuf[4][16 * 40];  // per-wave 16x32 P tile, stride 40

    int lane = threadIdx.x & 31;
    int wave = threadIdx.x >> 5;
    int col  = lane & 15;
    int half = lane >> 4;

    int bh = blockIdx.y;
    int b  = bh / H_;
    int h  = bh % H_;
    int kv = h / NREP_;
    int qr = blockIdx.x * 64 + wave * 16;

    // Q fragments (A layout): lane row = qr+col, K chunks {8h..,16+8h..}
    const __bf16* qrow = Q + (((size_t)(b * H_ + h)) * S_ + qr + col) * HD_;
    Frag qa0, qa1;
    qa0.q[0] = *(const uint4*)(qrow + 8 * half);
    qa0.q[1] = *(const uint4*)(qrow + 16 + 8 * half);
    qa1.q[0] = *(const uint4*)(qrow + 32 + 8 * half);
    qa1.q[1] = *(const uint4*)(qrow + 48 + 8 * half);

    const __bf16* kbase = Kc + ((size_t)(b * HKV_ + kv)) * (size_t)S_ * HD_;
    const __bf16* vbase = VT + ((size_t)(b * HKV_ + kv)) * (size_t)HD_ * S_;
    const int*    mrow  = amask + (size_t)b * S_;

    v8f acc[4];
#pragma unroll
    for (int f = 0; f < 4; ++f) acc[f] = vzero8();
    float rmax[8], rsum[8];
#pragma unroll
    for (int r = 0; r < 8; ++r) { rmax[r] = -__builtin_inff(); rsum[r] = 0.0f; }

    __bf16* pb = pbuf[wave];

    int kend = qr + 16;  // causal: need keys <= qr+15
    for (int kt = 0; kt < kend; kt += 32) {
        // prefetch next key tile (speculative, L0/L2 warm-up)
        if (kt + 32 < kend) {
            __builtin_prefetch(kbase + (size_t)(kt + 32 + lane) * HD_, 0, 1);
            __builtin_prefetch(vbase + (size_t)lane * S_ + kt + 32, 0, 1);
        }

        // ---- S = Q @ K^T for 32 keys (two 16-column fragments) ----
        v8f s[2];
#pragma unroll
        for (int j = 0; j < 2; ++j) {
            // B = K^T: lane's column = key row of K (row-major, contiguous d)
            const __bf16* krow = kbase + (size_t)(kt + 16 * j + col) * HD_;
            Frag kb0, kb1;
            kb0.q[0] = *(const uint4*)(krow + 8 * half);
            kb0.q[1] = *(const uint4*)(krow + 16 + 8 * half);
            kb1.q[0] = *(const uint4*)(krow + 32 + 8 * half);
            kb1.q[1] = *(const uint4*)(krow + 48 + 8 * half);
            v8f z = vzero8();
            z = wmma_bf16(qa0, kb0, z);   // d = 0..31
            z = wmma_bf16(qa1, kb1, z);   // d = 32..63
            s[j] = z;
        }

        // ---- scale + causal/attention mask + online softmax ----
        int key0 = kt + col;
        int key1 = kt + 16 + col;
        bool av0 = (mrow[key0] != 0);
        bool av1 = (mrow[key1] != 0);

        float p0[8], p1[8], tmax[8];
#pragma unroll
        for (int r = 0; r < 8; ++r) {
            int m = qr + r + 8 * half;
            float v0 = (av0 && key0 <= m) ? s[0][r] * SCALE_ : -__builtin_inff();
            float v1 = (av1 && key1 <= m) ? s[1][r] * SCALE_ : -__builtin_inff();
            p0[r] = v0;
            p1[r] = v1;
            tmax[r] = fmaxf(v0, v1);
        }
#pragma unroll
        for (int off = 1; off < 16; off <<= 1)
#pragma unroll
            for (int r = 0; r < 8; ++r)
                tmax[r] = fmaxf(tmax[r], __shfl_xor(tmax[r], off, 32));

        float corr[8];
#pragma unroll
        for (int r = 0; r < 8; ++r) {
            float nm = fmaxf(rmax[r], tmax[r]);
            corr[r] = __expf(rmax[r] - nm);
            rmax[r] = nm;
            p0[r] = __expf(p0[r] - nm);
            p1[r] = __expf(p1[r] - nm);
        }
        float tsum[8];
#pragma unroll
        for (int r = 0; r < 8; ++r) tsum[r] = p0[r] + p1[r];
#pragma unroll
        for (int off = 1; off < 16; off <<= 1)
#pragma unroll
            for (int r = 0; r < 8; ++r)
                tsum[r] += __shfl_xor(tsum[r], off, 32);
#pragma unroll
        for (int r = 0; r < 8; ++r) rsum[r] = rsum[r] * corr[r] + tsum[r];
#pragma unroll
        for (int f = 0; f < 4; ++f)
#pragma unroll
            for (int r = 0; r < 8; ++r) acc[f][r] *= corr[r];

        // ---- P: D layout -> LDS -> A layout (wave-private tile) ----
#pragma unroll
        for (int r = 0; r < 8; ++r) {
            int row = r + 8 * half;
            pb[row * 40 + col]      = (__bf16)p0[r];
            pb[row * 40 + 16 + col] = (__bf16)p1[r];
        }
        asm volatile("s_wait_dscnt 0" ::: "memory");

        Frag pf;
        const __bf16* prow = pb + col * 40;
        pf.q[0] = *(const uint4*)(prow + 8 * half);
        pf.q[1] = *(const uint4*)(prow + 16 + 8 * half);

        // ---- acc += P(16x32) @ V(32xHD) ; B fragment from VT rows ----
#pragma unroll
        for (int f = 0; f < 4; ++f) {
            const __bf16* vrow = vbase + (size_t)(16 * f + col) * S_ + kt + 8 * half;
            Frag vf;
            vf.q[0] = *(const uint4*)(vrow);
            vf.q[1] = *(const uint4*)(vrow + 16);
            acc[f] = wmma_bf16(pf, vf, acc[f]);
        }
        asm volatile("" ::: "memory");  // keep next tile's LDS stores after these reads
    }

    // ---- normalize and write O[B,S,H,HD] ----
#pragma unroll
    for (int f = 0; f < 4; ++f)
#pragma unroll
        for (int r = 0; r < 8; ++r) {
            float o = acc[f][r] / rsum[r];
            O[(((size_t)b * S_ + qr + r + 8 * half) * H_ + h) * HD_ + 16 * f + col] =
                (__bf16)o;
        }
}

// ---------------------------------------------------------------------------
// Host-side launcher
// ---------------------------------------------------------------------------
extern "C" void kernel_launch(void* const* d_in, const int* in_sizes, int n_in,
                              void* d_out, int out_size, void* d_ws, size_t ws_size,
                              hipStream_t stream) {
    const float* hs    = (const float*)d_in[0];
    const int*   amask = (const int*)d_in[1];
    const float* Wq    = (const float*)d_in[2];
    const float* Wk    = (const float*)d_in[3];
    const float* Wv    = (const float*)d_in[4];
    const float* Wo    = (const float*)d_in[5];
    float*       out   = (float*)d_out;

    char*  ws  = (char*)d_ws;
    size_t off = 0;
    auto carve = [&](size_t bytes) -> void* {
        void* p = ws + off;
        off += (bytes + 255) & ~(size_t)255;
        return p;
    };

    __bf16* hb    = (__bf16*)carve((size_t)TOK_ * D_ * 2);          // hidden bf16
    __bf16* WqT   = (__bf16*)carve((size_t)D_ * D_ * 2);            // [N=2048, K=2048]
    __bf16* WkT   = (__bf16*)carve((size_t)(HKV_ * HD_) * D_ * 2);  // [512, 2048]
    __bf16* WvT   = (__bf16*)carve((size_t)(HKV_ * HD_) * D_ * 2);
    __bf16* WoT   = (__bf16*)carve((size_t)D_ * D_ * 2);
    float*  qf    = (float*)carve((size_t)TOK_ * D_ * 4);                 // [tok, H*HD]
    float*  kf    = (float*)carve((size_t)TOK_ * HKV_ * HD_ * 4);
    float*  vf    = (float*)carve((size_t)TOK_ * HKV_ * HD_ * 4);
    __bf16* qb    = (__bf16*)carve((size_t)B_ * H_ * S_ * HD_ * 2);       // [B,H,S,HD]
    __bf16* kb    = (__bf16*)carve((size_t)B_ * HKV_ * S_ * HD_ * 2);     // [B,HKV,S,HD]
    __bf16* vtb   = (__bf16*)carve((size_t)B_ * HKV_ * HD_ * S_ * 2);     // [B,HKV,HD,S]
    __bf16* attnb = (__bf16*)carve((size_t)TOK_ * D_ * 2);                // [tok, H*HD]

    // 1) casts / weight transposes
    {
        long n = (long)TOK_ * D_;
        cast_f32_bf16<<<(n + 255) / 256, 256, 0, stream>>>(hs, hb, n);
    }
    transpose_cast<<<dim3(D_ / 32, D_ / 32), dim3(32, 8), 0, stream>>>(Wq, WqT, D_, D_);
    transpose_cast<<<dim3((HKV_ * HD_) / 32, D_ / 32), dim3(32, 8), 0, stream>>>(Wk, WkT, D_, HKV_ * HD_);
    transpose_cast<<<dim3((HKV_ * HD_) / 32, D_ / 32), dim3(32, 8), 0, stream>>>(Wv, WvT, D_, HKV_ * HD_);
    transpose_cast<<<dim3(D_ / 32, D_ / 32), dim3(32, 8), 0, stream>>>(Wo, WoT, D_, D_);

    // 2) QKV projections (WMMA GEMM, async-LDS staged B)
    gemm_bf16_f32<<<dim3((H_ * HD_) / 128, TOK_ / 128), 256, 0, stream>>>(
        hb, WqT, qf, TOK_, H_ * HD_, D_);
    gemm_bf16_f32<<<dim3((HKV_ * HD_) / 128, TOK_ / 128), 256, 0, stream>>>(
        hb, WkT, kf, TOK_, HKV_ * HD_, D_);
    gemm_bf16_f32<<<dim3((HKV_ * HD_) / 128, TOK_ / 128), 256, 0, stream>>>(
        hb, WvT, vf, TOK_, HKV_ * HD_, D_);

    // 3) RoPE + layout
    {
        long nq = (long)B_ * S_ * H_ * (HD_ / 2);
        rope_cast<<<(nq + 255) / 256, 256, 0, stream>>>(qf, qb, H_);
        long nk = (long)B_ * S_ * HKV_ * (HD_ / 2);
        rope_cast<<<(nk + 255) / 256, 256, 0, stream>>>(kf, kb, HKV_);
    }
    v_transpose_cast<<<dim3(S_ / 32, HD_ / 32, B_ * HKV_), dim3(32, 8), 0, stream>>>(vf, vtb);

    // 4) flash attention
    flash_attn<<<dim3(S_ / 64, B_ * H_), 128, 0, stream>>>(qb, kb, vtb, amask, attnb);

    // 5) output projection -> d_out (f32)
    gemm_bf16_f32<<<dim3(D_ / 128, TOK_ / 128), 256, 0, stream>>>(
        attnb, WoT, out, TOK_, D_, D_);

    (void)in_sizes; (void)n_in; (void)out_size; (void)ws_size;
}
